// SoftAttentionKNNGraph_11123965296912
// MI455X (gfx1250) — matrix operand order repaired and checked by reading, not compiled, not run on hardware
//
#include <hip/hip_runtime.h>

// Problem constants (fixed by the reference).
#define NROW 8192
#define KD   512
#define KTOP 5

// Wave tile: MI x NJ WMMA tiles (each 16x16). 2x4 = 32 rows x 64 cols.
#define MI 2
#define NJ 4

typedef __attribute__((ext_vector_type(16))) __bf16 v16bf;
typedef __attribute__((ext_vector_type(8)))  __bf16 v8bf;
typedef __attribute__((ext_vector_type(8)))  float  v8f;

__device__ __forceinline__ v16bf cat8(v8bf a, v8bf b) {
  v16bf r;
#pragma unroll
  for (int i = 0; i < 8; ++i) { r[i] = a[i]; r[i + 8] = b[i]; }
  return r;
}

// ---------------------------------------------------------------------------
// Kernel 1: row L2-normalize and split each normalized fp32 into bf16 hi+lo.
// One 256-thread block per row (512 elements -> 2 per thread).
// ---------------------------------------------------------------------------
__global__ __launch_bounds__(256) void k_normalize(const float* __restrict__ X,
                                                   __bf16* __restrict__ Xhi,
                                                   __bf16* __restrict__ Xlo) {
  const int row = blockIdx.x;
  const float* x = X + (size_t)row * KD;

  float vals[2];
  float ss = 0.f;
#pragma unroll
  for (int i = 0; i < 2; ++i) {
    vals[i] = x[threadIdx.x + i * 256];
    ss += vals[i] * vals[i];
  }
  // wave32 tree reduce
#pragma unroll
  for (int off = 16; off > 0; off >>= 1) ss += __shfl_down(ss, off);

  __shared__ float red[8];
  __shared__ float s_inv;
  const int wave = threadIdx.x >> 5, lane = threadIdx.x & 31;
  if (lane == 0) red[wave] = ss;
  __syncthreads();
  if (threadIdx.x == 0) {
    float t = 0.f;
#pragma unroll
    for (int i = 0; i < 8; ++i) t += red[i];
    float nrm = fmaxf(sqrtf(t), 1e-12f);  // eps matches reference
    s_inv = 1.0f / nrm;
  }
  __syncthreads();
  const float inv = s_inv;
#pragma unroll
  for (int i = 0; i < 2; ++i) {
    const int c = threadIdx.x + i * 256;
    const float xn = vals[i] * inv;
    const __bf16 h = (__bf16)xn;
    const __bf16 l = (__bf16)(xn - (float)h);
    Xhi[(size_t)row * KD + c] = h;
    Xlo[(size_t)row * KD + c] = l;
  }
}

// ---------------------------------------------------------------------------
// Kernel 2: sim = Xn * Xn^T via split-bf16 WMMA (hi*hi + hi*lo + lo*hi).
// Block tile 64 rows x 256 cols; 8 waves (wave32) in a 2x4 grid; each wave
// owns a 32x64 tile = 2x4 WMMA tiles -> 24 v_wmma per k-step of 32.
// Operand planes total 16 MB -> fully L2-resident; fragments are loaded
// straight from global (L2) as b128 clauses, no LDS staging needed.
//
// A fragment (16x32 bf16): lane l (r=l&15, half=l>>4) holds row
//   m0+r; v[0:7] = K k0+half*16 .. +15
//   -> one contiguous 32B load per lane.
// B fragment (32x16 bf16): lane holds column n0+(l&15) (a row of Xn, since
//   B = Xn^T); v[0:7] = K k0+half*16 .. +15 -> one contiguous 32B load.
// ---------------------------------------------------------------------------
__global__ __launch_bounds__(256) void k_gemm(const __bf16* __restrict__ Xhi,
                                              const __bf16* __restrict__ Xlo,
                                              float* __restrict__ C) {
  const int wave = threadIdx.x >> 5;
  const int lane = threadIdx.x & 31;
  const int wr   = wave >> 2;   // 0..1
  const int wc   = wave & 3;    // 0..3
  const int half = lane >> 4;   // 0/1
  const int l16  = lane & 15;

  const int m0 = blockIdx.y * (64) + wr * (MI * 16);        // 32-row wave tile
  const int n0 = blockIdx.x * (256) + wc * (NJ * 16);       // 64-col wave tile

  v8f acc[MI][NJ] = {};

  for (int k0 = 0; k0 < KD; k0 += 32) {
    v16bf ahi[MI], alo[MI], bhi[NJ], blo[NJ];
#pragma unroll
    for (int i = 0; i < MI; ++i) {
      const size_t rbase = (size_t)(m0 + i * 16 + l16) * KD;
      const __bf16* ph = Xhi + rbase + k0 + half * 16;
      const __bf16* pl = Xlo + rbase + k0 + half * 16;
      ahi[i] = *(const v16bf*)ph;
      alo[i] = *(const v16bf*)pl;
    }
#pragma unroll
    for (int j = 0; j < NJ; ++j) {
      const size_t cbase = (size_t)(n0 + j * 16 + l16) * KD;
      bhi[j] = *(const v16bf*)(Xhi + cbase + k0 + half * 16);
      blo[j] = *(const v16bf*)(Xlo + cbase + k0 + half * 16);
    }
#pragma unroll
    for (int i = 0; i < MI; ++i)
#pragma unroll
      for (int j = 0; j < NJ; ++j) {
        acc[i][j] = __builtin_amdgcn_wmma_f32_16x16x32_bf16(
            false, ahi[i], false, bhi[j], (short)0, acc[i][j], false, false);
        acc[i][j] = __builtin_amdgcn_wmma_f32_16x16x32_bf16(
            false, ahi[i], false, blo[j], (short)0, acc[i][j], false, false);
        acc[i][j] = __builtin_amdgcn_wmma_f32_16x16x32_bf16(
            false, alo[i], false, bhi[j], (short)0, acc[i][j], false, false);
      }
  }

  // C/D layout: element (vgpr v, lane l): M = v + (l>>4)*8, N = l&15.
#pragma unroll
  for (int i = 0; i < MI; ++i)
#pragma unroll
    for (int j = 0; j < NJ; ++j)
#pragma unroll
      for (int v = 0; v < 8; ++v) {
        const int r = m0 + i * 16 + half * 8 + v;
        const int c = n0 + j * 16 + l16;
        C[(size_t)r * NROW + c] = acc[i][j][v];
      }
}

// ---------------------------------------------------------------------------
// Kernel 3: in-place per-row top-5 + softmax(tau=0.1). One block per row.
// Per-thread sorted top-5 (ties -> lower index, matching lax.top_k), LDS
// merge tree across 256 threads, then coalesced rewrite of the row.
// ---------------------------------------------------------------------------
__global__ __launch_bounds__(256) void k_topk_softmax(float* __restrict__ S) {
  const int row = blockIdx.x;
  float* r = S + (size_t)row * NROW;
  const int t = threadIdx.x;

  float v[KTOP];
  int   id[KTOP];
#pragma unroll
  for (int j = 0; j < KTOP; ++j) { v[j] = -3.4e38f; id[j] = 0x7fffffff; }

  for (int c = t; c < NROW; c += 256) {
    const float x = r[c];
    if (x > v[KTOP - 1] || (x == v[KTOP - 1] && c < id[KTOP - 1])) {
      int p = KTOP - 1;
      while (p > 0 && (x > v[p - 1] || (x == v[p - 1] && c < id[p - 1]))) {
        v[p] = v[p - 1]; id[p] = id[p - 1]; --p;
      }
      v[p] = x; id[p] = c;
    }
  }

  __shared__ float sv[256 * KTOP];
  __shared__ int   si[256 * KTOP];
#pragma unroll
  for (int j = 0; j < KTOP; ++j) { sv[t * KTOP + j] = v[j]; si[t * KTOP + j] = id[j]; }
  __syncthreads();

  for (int s = 128; s >= 1; s >>= 1) {
    if (t < s) {
      float* av = &sv[t * KTOP];
      int*   ai = &si[t * KTOP];
      const float* bv = &sv[(t + s) * KTOP];
      const int*   bi = &si[(t + s) * KTOP];
      float mv[KTOP]; int mi[KTOP];
      int p = 0, q = 0;
#pragma unroll
      for (int o = 0; o < KTOP; ++o) {
        const bool ta = (av[p] > bv[q]) || (av[p] == bv[q] && ai[p] < bi[q]);
        if (ta) { mv[o] = av[p]; mi[o] = ai[p]; ++p; }
        else    { mv[o] = bv[q]; mi[o] = bi[q]; ++q; }
      }
#pragma unroll
      for (int o = 0; o < KTOP; ++o) { av[o] = mv[o]; ai[o] = mi[o]; }
    }
    __syncthreads();
  }

  __shared__ float pv[KTOP];
  __shared__ int   pid[KTOP];
  if (t == 0) {
    const float m = sv[0];
    float e[KTOP], sum = 0.f;
#pragma unroll
    for (int j = 0; j < KTOP; ++j) { e[j] = __expf((sv[j] - m) * 10.0f); sum += e[j]; }
    const float inv = 1.0f / sum;
#pragma unroll
    for (int j = 0; j < KTOP; ++j) { pv[j] = e[j] * inv; pid[j] = si[j]; }
  }
  __syncthreads();

  // Non-top-k entries: exp((-1e9 - max)/0.1) underflows to exactly 0 in fp32.
  for (int c = t; c < NROW; c += 256) {
    float out = 0.f;
#pragma unroll
    for (int j = 0; j < KTOP; ++j) out = (c == pid[j]) ? pv[j] : out;
    r[c] = out;
  }
}

// ---------------------------------------------------------------------------
extern "C" void kernel_launch(void* const* d_in, const int* in_sizes, int n_in,
                              void* d_out, int out_size, void* d_ws, size_t ws_size,
                              hipStream_t stream) {
  (void)in_sizes; (void)n_in; (void)out_size; (void)ws_size;
  const float* X = (const float*)d_in[0];
  float* S = (float*)d_out;

  // Workspace: bf16 hi plane + bf16 lo plane of normalized X (16 MB total).
  __bf16* Xhi = (__bf16*)d_ws;
  __bf16* Xlo = Xhi + (size_t)NROW * KD;

  k_normalize<<<NROW, 256, 0, stream>>>(X, Xhi, Xlo);

  dim3 g(NROW / 256, NROW / 64);  // (col tiles, row tiles)
  k_gemm<<<g, 256, 0, stream>>>(Xhi, Xlo, S);

  k_topk_softmax<<<NROW, 256, 0, stream>>>(S);
}
